// MultiheadReliableAttention_6743098655129
// MI455X (gfx1250) — compile-verified, matrix operand order checked by reading, not run on hardware
//
#include <hip/hip_runtime.h>
#include <math.h>

// ---------------- problem constants ----------------
#define LSEQ 2048
#define BSZ  2
#define EDIM 1024
#define NH   16
#define HD   64
#define TOPK 64
#define NROW (BSZ*LSEQ)      // 4096 rows for the big GEMMs
#define NBH  (BSZ*NH)        // 32 (b,h) pairs

typedef __attribute__((ext_vector_type(16))) __bf16 v16bf;
typedef __attribute__((ext_vector_type(8)))  __bf16 v8bf;
typedef __attribute__((ext_vector_type(8)))  float  v8f;

#ifndef __has_builtin
#define __has_builtin(x) 0
#endif
#if __has_builtin(__builtin_amdgcn_global_load_async_to_lds_b128)
#define HAVE_ASYNC_LDS 1
#else
#define HAVE_ASYNC_LDS 0
#endif
#if __has_builtin(__builtin_amdgcn_s_wait_asynccnt)
#define HAVE_WAIT_ASYNC 1
#else
#define HAVE_WAIT_ASYNC 0
#endif

__device__ __forceinline__ __bf16 f32_to_bf16(float f) {
  unsigned int u = __builtin_bit_cast(unsigned int, f);
  unsigned int lsb = (u >> 16) & 1u;
  u += 0x7fffu + lsb;                      // round-to-nearest-even
  unsigned short h = (unsigned short)(u >> 16);
  return __builtin_bit_cast(__bf16, h);
}

// 16B global -> LDS copy: async (CDNA5 GLOBAL_LOAD_ASYNC_TO_LDS_B128) if available
#if HAVE_ASYNC_LDS
typedef __attribute__((__vector_size__(4 * sizeof(int)))) int b128_t;
typedef __attribute__((address_space(1))) b128_t g_b128_t;   // global
typedef __attribute__((address_space(3))) b128_t l_b128_t;   // LDS
#endif

__device__ __forceinline__ void copy16_g2l(const __bf16* g, __bf16* l) {
#if HAVE_ASYNC_LDS
  __builtin_amdgcn_global_load_async_to_lds_b128((g_b128_t*)g, (l_b128_t*)l, 0, 0);
#else
  *(v8bf*)l = *(const v8bf*)g;
#endif
}

__device__ __forceinline__ void wait_async_copies() {
#if HAVE_ASYNC_LDS
#if HAVE_WAIT_ASYNC
  __builtin_amdgcn_s_wait_asynccnt(0);
#else
  asm volatile("s_wait_asynccnt 0x0" ::: "memory");
#endif
#endif
}

__device__ __forceinline__ v16bf load_frag(const __bf16* p) {
  v8bf lo = *(const v8bf*)p;
  v8bf hi = *(const v8bf*)(p + 16);
  v16bf r;
#pragma unroll
  for (int i = 0; i < 8; ++i) { r[i] = lo[i]; r[8 + i] = hi[i]; }
  return r;
}

// ---------------- pack kernels ----------------
// query [L,B,E] f32  ->  xbf [B,L,E] bf16
__global__ __launch_bounds__(256) void pack_x(const float* __restrict__ q,
                                              __bf16* __restrict__ xbf) {
  int i = blockIdx.x * 256 + threadIdx.x;      // output linear index over B*L*E
  int e = i & (EDIM - 1);
  int r = i >> 10;                             // b*L + l
  int b = r >> 11;
  int l = r & (LSEQ - 1);
  xbf[i] = f32_to_bf16(q[((size_t)(l * BSZ + b) << 10) + e]);
}

__global__ __launch_bounds__(256) void pack_w(const float* __restrict__ w,
                                              __bf16* __restrict__ wbf) {
  int i = blockIdx.x * 256 + threadIdx.x;      // E*E
  wbf[i] = f32_to_bf16(w[i]);
}

// ---------------- WMMA GEMM ----------------
// C[n,j] = sum_e A[n,e] * W[j,e] + bias[j]
// Macro tile 128(M) x 64(N); 8 waves, each wave a 32x32 tile (4 wmma / K-step).
// Double-buffered LDS with async global->LDS tile fills.
// mode 0: write f32 to outQKV laid out [B,H,L,HD]  (n=b*L+l, j=h*64+d)
// mode 1: write outFinal[l,b,j] = C + resid[l,b,j]
__global__ __launch_bounds__(256)
void gemm_bf16(const __bf16* __restrict__ A,    // [NROW,E] bf16 row-major
               const __bf16* __restrict__ W,    // [E,E]   bf16 row-major (W[j][e])
               const float*  __restrict__ bias, // [E]
               float* __restrict__ outQKV,
               float* __restrict__ outFinal,
               const float* __restrict__ resid,
               int mode) {
  constexpr int LDT = 40;                       // padded LDS row stride (bf16 elems)
  __shared__ __bf16 At[2][128 * LDT];           // 2 x 10 KB
  __shared__ __bf16 Bt[2][64 * LDT];            // 2 x  5 KB

  const int tid    = threadIdx.x;
  const int blockM = blockIdx.x * 128;
  const int blockN = blockIdx.y * 64;

  const int w    = tid >> 5;                    // wave 0..7
  const int lid  = tid & 31;
  const int wm   = w & 3;                       // M 32-tile 0..3
  const int wn   = w >> 2;                      // N 32-tile 0..1
  const int half = lid >> 4;                    // 0/1
  const int rrow = lid & 15;

  // tile-fill mapping: 16B chunks; A has 512 chunks (2/thread), B has 256 (1/thread)
  const int frow = tid >> 2;                    // 0..63
  const int fcol = (tid & 3) * 8;               // 0,8,16,24

  auto fill = [&](int buf, int k0) {
    copy16_g2l(&A[(size_t)(blockM + frow) * EDIM + k0 + fcol],
               &At[buf][frow * LDT + fcol]);
    copy16_g2l(&A[(size_t)(blockM + 64 + frow) * EDIM + k0 + fcol],
               &At[buf][(64 + frow) * LDT + fcol]);
    copy16_g2l(&W[(size_t)(blockN + frow) * EDIM + k0 + fcol],
               &Bt[buf][frow * LDT + fcol]);
  };

  v8f c00 = {}, c01 = {}, c10 = {}, c11 = {};

  fill(0, 0);
  wait_async_copies();
  __syncthreads();

  for (int k0 = 0, t = 0; k0 < EDIM; k0 += 32, ++t) {
    const int cur = t & 1;
    if (k0 + 32 < EDIM) fill(cur ^ 1, k0 + 32);

    // fragments per ISA 16-bit A/B layout:
    // lane(0-15): K = 0..7 & 16..23 ; lane(16-31): K = 8..15 & 24..31
    v16bf a0 = load_frag(&At[cur][(wm * 32 + rrow) * LDT + half * 8]);
    v16bf a1 = load_frag(&At[cur][(wm * 32 + 16 + rrow) * LDT + half * 8]);
    v16bf b0 = load_frag(&Bt[cur][(wn * 32 + rrow) * LDT + half * 8]);
    v16bf b1 = load_frag(&Bt[cur][(wn * 32 + 16 + rrow) * LDT + half * 8]);

    c00 = __builtin_amdgcn_wmma_f32_16x16x32_bf16(false, a0, false, b0, (short)0, c00, false, false);
    c01 = __builtin_amdgcn_wmma_f32_16x16x32_bf16(false, a0, false, b1, (short)0, c01, false, false);
    c10 = __builtin_amdgcn_wmma_f32_16x16x32_bf16(false, a1, false, b0, (short)0, c10, false, false);
    c11 = __builtin_amdgcn_wmma_f32_16x16x32_bf16(false, a1, false, b1, (short)0, c11, false, false);

    wait_async_copies();     // next tile fully in LDS (per-wave) ...
    __syncthreads();         // ... and across the workgroup
  }

  // C/D layout: c[r] -> M = half*8 + r, N = rrow (per 16x16 subtile)
  const int n0 = blockN + wn * 32 + rrow;
  const int n1 = n0 + 16;
  const float bi0 = bias[n0];
  const float bi1 = bias[n1];
#pragma unroll
  for (int r = 0; r < 8; ++r) {
#pragma unroll
    for (int sm = 0; sm < 2; ++sm) {
      int m = blockM + wm * 32 + sm * 16 + half * 8 + r;
      float v0 = (sm == 0 ? c00[r] : c10[r]) + bi0;
      float v1 = (sm == 0 ? c01[r] : c11[r]) + bi1;
      int b = m >> 11;
      int l = m & (LSEQ - 1);
      if (mode == 0) {
        int h0 = n0 >> 6, d0 = n0 & 63;
        int h1 = n1 >> 6, d1 = n1 & 63;
        outQKV[(((size_t)(b * NH + h0) * LSEQ + l) << 6) + d0] = v0;
        outQKV[(((size_t)(b * NH + h1) * LSEQ + l) << 6) + d1] = v1;
      } else {
        size_t base = ((size_t)l * BSZ + b) * EDIM;
        outFinal[base + n0] = v0 + resid[base + n0];
        outFinal[base + n1] = v1 + resid[base + n1];
      }
    }
  }
}

// ---------------- qsum + per-key score ----------------
// score[bh,l] = scale * K[bh,l,:] . (sum_q Q[bh,q,:])
__global__ __launch_bounds__(256) void qsum_score(const float* __restrict__ Qf,
                                                  const float* __restrict__ Kf,
                                                  float* __restrict__ scores) {
  const int bh = blockIdx.x;
  const float* Qh = Qf + (size_t)bh * LSEQ * HD;
  const float* Kh = Kf + (size_t)bh * LSEQ * HD;
  __shared__ float part[256];
  __shared__ float qs[HD];
  const int tid = threadIdx.x;
  const int d = tid & 63, seg = tid >> 6;
  float acc = 0.f;
  for (int l = seg; l < LSEQ; l += 4) acc += Qh[(size_t)l * HD + d];
  part[tid] = acc;
  __syncthreads();
  if (tid < HD) qs[tid] = part[tid] + part[tid + 64] + part[tid + 128] + part[tid + 192];
  __syncthreads();
  const float scale = 0.125f;   // HD^-0.5
  for (int l = tid; l < LSEQ; l += 256) {
    float s = 0.f;
#pragma unroll 8
    for (int dd = 0; dd < HD; ++dd) s += Kh[(size_t)l * HD + dd] * qs[dd];
    scores[(size_t)bh * LSEQ + l] = s * scale;
  }
}

// ---------------- top-k (iterative argmax) + gather selected K ----------------
__global__ __launch_bounds__(256) void topk_gather(const float* __restrict__ scores,
                                                   const float* __restrict__ Kf,
                                                   int* __restrict__ topidx,
                                                   float* __restrict__ selK) {
  const int bh = blockIdx.x;
  __shared__ float sc[LSEQ];
  __shared__ float rv[256];
  __shared__ int   ri[256];
  __shared__ int   chosen[TOPK];
  const int tid = threadIdx.x;
  for (int l = tid; l < LSEQ; l += 256) sc[l] = scores[(size_t)bh * LSEQ + l];
  __syncthreads();
  for (int it = 0; it < TOPK; ++it) {
    float bv = -INFINITY; int bi = LSEQ;
    for (int l = tid; l < LSEQ; l += 256) {
      float v = sc[l];
      if (v > bv || (v == bv && l < bi)) { bv = v; bi = l; }
    }
    rv[tid] = bv; ri[tid] = bi;
    __syncthreads();
    for (int off = 128; off > 0; off >>= 1) {
      if (tid < off) {
        float v2 = rv[tid + off]; int i2 = ri[tid + off];
        if (v2 > rv[tid] || (v2 == rv[tid] && i2 < ri[tid])) { rv[tid] = v2; ri[tid] = i2; }
      }
      __syncthreads();
    }
    if (tid == 0) { chosen[it] = ri[0]; sc[ri[0]] = -INFINITY; }
    __syncthreads();
  }
  if (tid < TOPK) topidx[bh * TOPK + tid] = chosen[tid];
  __syncthreads();
  for (int p = tid; p < TOPK * HD; p += 256) {
    int m = p >> 6, d = p & 63;
    selK[((size_t)bh * TOPK + m) * HD + d] =
        Kf[((size_t)bh * LSEQ + chosen[m]) * HD + d];
  }
}

// ---------------- softmax(X . selK^T * scale) -> sim[bh,l,m] ----------------
__global__ __launch_bounds__(256) void sim_softmax(const float* __restrict__ X,
                                                   const float* __restrict__ selK,
                                                   float* __restrict__ sim) {
  const int bh = blockIdx.x;
  const int l  = blockIdx.y * 256 + threadIdx.x;
  __shared__ float sk[TOPK * HD];
  for (int p = threadIdx.x; p < TOPK * HD; p += 256)
    sk[p] = selK[(size_t)bh * TOPK * HD + p];
  __syncthreads();
  const float* x = X + ((size_t)bh * LSEQ + l) * HD;
  float xr[HD];
#pragma unroll
  for (int d = 0; d < HD; ++d) xr[d] = x[d];
  float s[TOPK];
  float mx = -INFINITY;
  for (int m = 0; m < TOPK; ++m) {
    float a = 0.f;
#pragma unroll 8
    for (int d = 0; d < HD; ++d) a += xr[d] * sk[m * HD + d];
    a *= 0.125f;
    s[m] = a;
    mx = fmaxf(mx, a);
  }
  float sum = 0.f;
  for (int m = 0; m < TOPK; ++m) { s[m] = __expf(s[m] - mx); sum += s[m]; }
  const float inv = 1.f / sum;
  float* o = sim + ((size_t)bh * LSEQ + l) * TOPK;
  for (int m = 0; m < TOPK; ++m) o[m] = s[m] * inv;
}

// ---------------- tmp[bh,m,d] = sum_l simk[bh,l,m] * V[bh,l,d] ----------------
__global__ __launch_bounds__(256) void attn_tmp(const float* __restrict__ simk,
                                                const float* __restrict__ Vf,
                                                float* __restrict__ tmp) {
  const int bh = blockIdx.x;
  __shared__ float skc[64 * 64];   // [l_local][m]
  __shared__ float vc[64 * 64];    // [l_local][d]
  const int tid = threadIdx.x;
  float acc[16];
#pragma unroll
  for (int i = 0; i < 16; ++i) acc[i] = 0.f;
  for (int c = 0; c < LSEQ; c += 64) {
    for (int p = tid; p < 64 * 64; p += 256) {
      int ll = p >> 6, q = p & 63;
      skc[p] = simk[((size_t)bh * LSEQ + c + ll) * TOPK + q];
      vc[p]  = Vf[((size_t)bh * LSEQ + c + ll) * HD + q];
    }
    __syncthreads();
#pragma unroll
    for (int i = 0; i < 16; ++i) {
      int p = tid * 16 + i;
      int m = p >> 6, d = p & 63;
      float a = acc[i];
#pragma unroll 8
      for (int ll = 0; ll < 64; ++ll) a += skc[ll * 64 + m] * vc[ll * 64 + d];
      acc[i] = a;
    }
    __syncthreads();
  }
#pragma unroll
  for (int i = 0; i < 16; ++i) tmp[(size_t)bh * 4096 + tid * 16 + i] = acc[i];
}

// ---------------- out[b,l,h*64+d] = softmax(q.selK^T)[m] . tmp[m,d] (bf16) ----
__global__ __launch_bounds__(256) void attn_out(const float* __restrict__ Qf,
                                                const float* __restrict__ selK,
                                                const float* __restrict__ tmp,
                                                __bf16* __restrict__ attnbf) {
  const int bh = blockIdx.x;
  const int b = bh >> 4, h = bh & 15;
  const int l = blockIdx.y * 256 + threadIdx.x;
  __shared__ float sk[64 * 64];
  __shared__ float tp[64 * 64];
  for (int p = threadIdx.x; p < 64 * 64; p += 256) {
    sk[p] = selK[(size_t)bh * 4096 + p];
    tp[p] = tmp[(size_t)bh * 4096 + p];
  }
  __syncthreads();
  const float* q = Qf + ((size_t)bh * LSEQ + l) * HD;
  float qr[HD];
#pragma unroll
  for (int d = 0; d < HD; ++d) qr[d] = q[d];
  float s[TOPK];
  float mx = -INFINITY;
  for (int m = 0; m < TOPK; ++m) {
    float a = 0.f;
#pragma unroll 8
    for (int d = 0; d < HD; ++d) a += qr[d] * sk[m * 64 + d];
    a *= 0.125f;
    s[m] = a;
    mx = fmaxf(mx, a);
  }
  float sum = 0.f;
  for (int m = 0; m < TOPK; ++m) { s[m] = __expf(s[m] - mx); sum += s[m]; }
  const float inv = 1.f / sum;
  float o[HD];
#pragma unroll
  for (int d = 0; d < HD; ++d) o[d] = 0.f;
  for (int m = 0; m < TOPK; ++m) {
    float wgt = s[m] * inv;
#pragma unroll 8
    for (int d = 0; d < HD; ++d) o[d] += wgt * tp[m * 64 + d];
  }
  __bf16* dst = attnbf + ((size_t)(b * LSEQ + l)) * EDIM + h * HD;
  for (int d = 0; d < HD; ++d) dst[d] = f32_to_bf16(o[d]);
}

// ---------------- host-side launch ----------------
extern "C" void kernel_launch(void* const* d_in, const int* in_sizes, int n_in,
                              void* d_out, int out_size, void* d_ws, size_t ws_size,
                              hipStream_t stream) {
  (void)in_sizes; (void)n_in; (void)out_size; (void)ws_size;
  const float* query = (const float*)d_in[0];
  const float* q_w = (const float*)d_in[1];
  const float* q_b = (const float*)d_in[2];
  const float* k_w = (const float*)d_in[3];
  const float* k_b = (const float*)d_in[4];
  const float* v_w = (const float*)d_in[5];
  const float* v_b = (const float*)d_in[6];
  const float* o_w = (const float*)d_in[7];
  const float* o_b = (const float*)d_in[8];
  float* out = (float*)d_out;

  char* ws = (char*)d_ws;
  __bf16* xbf    = (__bf16*)(ws + 0);          //  8 MB  [B,L,E]
  __bf16* wq_bf  = (__bf16*)(ws + 8388608);    //  2 MB
  __bf16* wk_bf  = (__bf16*)(ws + 10485760);   //  2 MB
  __bf16* wv_bf  = (__bf16*)(ws + 12582912);   //  2 MB
  __bf16* wo_bf  = (__bf16*)(ws + 14680064);   //  2 MB
  float*  Qf     = (float*) (ws + 16777216);   // 16 MB  [B,H,L,HD]
  float*  Kf     = (float*) (ws + 33554432);   // 16 MB
  float*  Vf     = (float*) (ws + 50331648);   // 16 MB
  float*  scores = (float*) (ws + 67108864);   // 256 KB [BH,L]
  int*    topidx = (int*)   (ws + 67371008);   //   8 KB
  float*  selK   = (float*) (ws + 67379200);   // 512 KB [BH,64,64]
  float*  simk   = (float*) (ws + 67903488);   // 16 MB  [BH,L,64]
  float*  tmp    = (float*) (ws + 84680704);   // 512 KB [BH,64,64]
  __bf16* attnbf = (__bf16*)(ws + 85204992);   //  8 MB  [B,L,E]

  // 1) pack inputs to bf16
  pack_x<<<(BSZ * LSEQ * EDIM) / 256, 256, 0, stream>>>(query, xbf);
  pack_w<<<(EDIM * EDIM) / 256, 256, 0, stream>>>(q_w, wq_bf);
  pack_w<<<(EDIM * EDIM) / 256, 256, 0, stream>>>(k_w, wk_bf);
  pack_w<<<(EDIM * EDIM) / 256, 256, 0, stream>>>(v_w, wv_bf);
  pack_w<<<(EDIM * EDIM) / 256, 256, 0, stream>>>(o_w, wo_bf);

  // 2) Q/K/V projections via WMMA GEMM (128x64 macro tiles)
  dim3 ggrid(NROW / 128, EDIM / 64);
  gemm_bf16<<<ggrid, 256, 0, stream>>>(xbf, wq_bf, q_b, Qf, nullptr, nullptr, 0);
  gemm_bf16<<<ggrid, 256, 0, stream>>>(xbf, wk_bf, k_b, Kf, nullptr, nullptr, 0);
  gemm_bf16<<<ggrid, 256, 0, stream>>>(xbf, wv_bf, v_b, Vf, nullptr, nullptr, 0);

  // 3) per-key scores via qsum trick (sum over q of q.k == (sum q).k)
  qsum_score<<<NBH, 256, 0, stream>>>(Qf, Kf, scores);

  // 4) top-64 keys per head + gather
  topk_gather<<<NBH, 256, 0, stream>>>(scores, Kf, topidx, selK);

  // 5) sim_km = softmax(K . selK^T * scale)
  sim_softmax<<<dim3(NBH, LSEQ / 256), 256, 0, stream>>>(Kf, selK, simk);

  // 6) tmp = sim_km^T @ V   (64x64 per head)
  attn_tmp<<<NBH, 256, 0, stream>>>(simk, Vf, tmp);

  // 7) out_attn = softmax(Q . selK^T * scale) @ tmp  -> bf16 [B,L,E]
  attn_out<<<dim3(NBH, LSEQ / 256), 256, 0, stream>>>(Qf, selK, tmp, attnbf);

  // 8) final projection + bias + residual -> [L,B,E]
  gemm_bf16<<<ggrid, 256, 0, stream>>>(attnbf, wo_bf, o_b, nullptr, out, query, 1);
}
